// GlobalSumSAKELayer_13108240187515
// MI455X (gfx1250) — matrix operand order repaired and checked by reading, not compiled, not run on hardware
//
#include <hip/hip_runtime.h>
#include <hip/hip_bf16.h>

typedef __attribute__((ext_vector_type(16))) _Float16 v16h;
typedef __attribute__((ext_vector_type(8)))  _Float16 v8h;
typedef __attribute__((ext_vector_type(8)))  float    v8f;

#define NB   32     // graphs
#define NN   128    // nodes per graph
#define FDIM 64     // input features
#define HID  128
#define OUTD 64
#define TI   16     // i-rows handled per workgroup

// ---- LDS layout (bytes), manually overlaid ----
#define OFF_W2T  0                    // f16 [HID][HID], N-major (W2^T)    32768
#define OFF_A    32768                // f32 [NN][HID]  A = H@W1a + b1     65536
#define OFF_BV   98304                // f32 [TI][HID]  Bv = H_i @ W1b      8192
#define OFF_B1   106496               // f32 [HID]
#define OFF_B2   107008               // f32 [HID]
#define OFF_W10  107520               // f32 [HID]  (W1 row 0, the d^2 row)
#define OFF_X    108032               // f32 [NN][3]
#define OFF_OVL  109568
// setup-phase overlay:
#define OFF_W1AT OFF_OVL              // f16 [HID][FDIM] (W1a^T)           16384
#define OFF_W1BT (OFF_OVL + 16384)    // f16 [HID][FDIM] (W1b^T)           16384
#define OFF_H    (OFF_OVL + 32768)    // f16 [NN][FDIM]                    16384
// main-phase overlay (same region, after __syncthreads):
#define OFF_Z0   OFF_OVL              // f16 [16][HID] z1 buffer 0          4096
#define OFF_Z1B  (OFF_OVL + 4096)     // f16 [16][HID] z1 buffer 1          4096
#define OFF_RED  (OFF_OVL + 8192)     // f32 [8 waves][32 lanes]            1024
#define SMEM_BYTES (OFF_OVL + 49152)  // 158720 B  (<= 160KB -> 2 WGs/WGP)

__device__ __forceinline__ float siluf(float x) {
    return x / (1.0f + __expf(-x));
}

__device__ __forceinline__ v16h cat8(v8h lo, v8h hi) {
    return __builtin_shufflevector(lo, hi, 0,1,2,3,4,5,6,7,8,9,10,11,12,13,14,15);
}

// All 256 threads cooperatively produce one 16x128 z1 tile (f16, row-major ==
// WMMA-A fragment order). Thread -> (row, 8 contiguous k's): one b128 store.
__device__ __forceinline__ void gen_z1(_Float16* __restrict__ zbuf,
                                       const float* __restrict__ sA,
                                       const float* __restrict__ sBv,
                                       const float* __restrict__ sw10,
                                       const float* __restrict__ sX,
                                       int tid, int i0, int t)
{
    const int r  = tid >> 4;            // tile row 0..15
    const int kb = (tid & 15) * 8;      // k base 0..120
    const int ii = t >> 3;              // local i (0..15)
    const int jt = t & 7;               // j-tile
    const int gi = i0 + ii;
    const int gj = jt * 16 + r;
    const float d0 = sX[gi * 3 + 0] - sX[gj * 3 + 0];
    const float d1 = sX[gi * 3 + 1] - sX[gj * 3 + 1];
    const float d2 = sX[gi * 3 + 2] - sX[gj * 3 + 2];
    const float dd = d0 * d0 + d1 * d1 + d2 * d2;

    const float4 a0 = *(const float4*)(sA  + gj * HID + kb);
    const float4 a1 = *(const float4*)(sA  + gj * HID + kb + 4);
    const float4 v0 = *(const float4*)(sBv + ii * HID + kb);
    const float4 v1 = *(const float4*)(sBv + ii * HID + kb + 4);
    const float4 w0 = *(const float4*)(sw10 + kb);
    const float4 w1 = *(const float4*)(sw10 + kb + 4);

    v8h z;
    z[0] = (_Float16)siluf(a0.x + v0.x + dd * w0.x);
    z[1] = (_Float16)siluf(a0.y + v0.y + dd * w0.y);
    z[2] = (_Float16)siluf(a0.z + v0.z + dd * w0.z);
    z[3] = (_Float16)siluf(a0.w + v0.w + dd * w0.w);
    z[4] = (_Float16)siluf(a1.x + v1.x + dd * w1.x);
    z[5] = (_Float16)siluf(a1.y + v1.y + dd * w1.y);
    z[6] = (_Float16)siluf(a1.z + v1.z + dd * w1.z);
    z[7] = (_Float16)siluf(a1.w + v1.w + dd * w1.w);
    *(v8h*)(zbuf + r * HID + kb) = z;
}

// One workgroup = (graph b, i-tile of 16 nodes). 8 waves, wave32.
// Each wave permanently owns one 16-wide N-tile of the layer-2 GEMM, keeping
// its W2 fragments register-resident for the entire kernel.
__global__ void __launch_bounds__(256)
sake_main_kernel(const float* __restrict__ feat, const float* __restrict__ coord,
                 const float* __restrict__ W1,   const float* __restrict__ b1,
                 const float* __restrict__ W2,   const float* __restrict__ b2,
                 float* __restrict__ partial)
{
    extern __shared__ unsigned char smem[];
    _Float16* sW2T  = (_Float16*)(smem + OFF_W2T);
    float*    sA    = (float*)(smem + OFF_A);
    float*    sBv   = (float*)(smem + OFF_BV);
    float*    sb1   = (float*)(smem + OFF_B1);
    float*    sb2   = (float*)(smem + OFF_B2);
    float*    sw10  = (float*)(smem + OFF_W10);
    float*    sX    = (float*)(smem + OFF_X);
    _Float16* sW1aT = (_Float16*)(smem + OFF_W1AT);
    _Float16* sW1bT = (_Float16*)(smem + OFF_W1BT);
    _Float16* sH    = (_Float16*)(smem + OFF_H);
    float*    sRed  = (float*)(smem + OFF_RED);
    _Float16* zb0   = (_Float16*)(smem + OFF_Z0);
    _Float16* zb1   = (_Float16*)(smem + OFF_Z1B);

    const int tid  = threadIdx.x;
    const int wave = tid >> 5;
    const int lane = tid & 31;
    const int lr   = lane & 15;          // row/col index inside a 16-wide tile
    const int hi   = (lane >> 4) & 1;    // lane half (K-offset selector)

    const int b  = blockIdx.x >> 3;      // graph
    const int it = blockIdx.x & 7;       // i-tile
    const int i0 = it * TI;

    // ---------------- cooperative global -> LDS (+ f32->f16 convert) --------
    for (int e = tid; e < HID * HID; e += 256) {
        int k = e >> 7, n = e & 127;
        sW2T[n * HID + k] = (_Float16)W2[e];                 // W2^T, N-major
    }
    for (int e = tid; e < FDIM * HID; e += 256) {
        int f = e >> 7, n = e & 127;
        sW1aT[n * FDIM + f] = (_Float16)W1[(1 + f) * HID + n];
        sW1bT[n * FDIM + f] = (_Float16)W1[(1 + FDIM + f) * HID + n];
    }
    for (int e = tid; e < NN * FDIM; e += 256)
        sH[e] = (_Float16)feat[b * NN * FDIM + e];
    for (int e = tid; e < NN * 3; e += 256)
        sX[e] = coord[b * NN * 3 + e];
    if (tid < HID) {
        sb1[tid]  = b1[tid];
        sb2[tid]  = b2[tid];
        sw10[tid] = W1[tid];             // W1 row 0 (multiplies d^2)
    }
    __syncthreads();

    // ---- load this wave's permanent W2 fragments (one N-tile, 4 K-steps) ---
    v16h breg[4];
    {
        const int n = wave * 16 + lr;
        #pragma unroll
        for (int ks = 0; ks < 4; ++ks) {
            const int kb = ks * 32 + (hi ? 16 : 0);
            v8h blo = *(const v8h*)(sW2T + n * HID + kb);
            v8h bhi = *(const v8h*)(sW2T + n * HID + kb + 8);
            breg[ks] = cat8(blo, bhi);
        }
    }

    // ---------------- setup GEMMs via WMMA: A = H@W1a + b1 ------------------
    {
        const int mt = wave;             // each wave owns one 16-row j-tile
        for (int nt = 0; nt < 8; ++nt) {
            const float bias = sb1[nt * 16 + lr];
            v8f acc;
            #pragma unroll
            for (int v = 0; v < 8; ++v) acc[v] = bias;       // fold b1 via C
            #pragma unroll
            for (int ks = 0; ks < 2; ++ks) {
                const int row = mt * 16 + lr;
                const int ka  = ks * 32 + (hi ? 16 : 0);
                v8h alo = *(const v8h*)(sH + row * FDIM + ka);
                v8h ahi = *(const v8h*)(sH + row * FDIM + ka + 8);
                const int n  = nt * 16 + lr;
                const int kb = ks * 32 + (hi ? 16 : 0);
                v8h blo = *(const v8h*)(sW1aT + n * FDIM + kb);
                v8h bhi = *(const v8h*)(sW1aT + n * FDIM + kb + 8);
                acc = __builtin_amdgcn_wmma_f32_16x16x32_f16(
                        false, cat8(alo, ahi), false, cat8(blo, bhi),
                        (short)0, acc, false, false);
            }
            #pragma unroll
            for (int v = 0; v < 8; ++v)
                sA[(mt * 16 + v + hi * 8) * HID + nt * 16 + lr] = acc[v];
        }
        // Bv for this workgroup's 16 i-rows (one M-tile): wave owns one N-tile
        {
            const int nt = wave;
            v8f acc = {};
            #pragma unroll
            for (int ks = 0; ks < 2; ++ks) {
                const int row = i0 + lr;
                const int ka  = ks * 32 + (hi ? 16 : 0);
                v8h alo = *(const v8h*)(sH + row * FDIM + ka);
                v8h ahi = *(const v8h*)(sH + row * FDIM + ka + 8);
                const int n  = nt * 16 + lr;
                const int kb = ks * 32 + (hi ? 16 : 0);
                v8h blo = *(const v8h*)(sW1bT + n * FDIM + kb);
                v8h bhi = *(const v8h*)(sW1bT + n * FDIM + kb + 8);
                acc = __builtin_amdgcn_wmma_f32_16x16x32_f16(
                        false, cat8(alo, ahi), false, cat8(blo, bhi),
                        (short)0, acc, false, false);
            }
            #pragma unroll
            for (int v = 0; v < 8; ++v)
                sBv[(v + hi * 8) * HID + nt * 16 + lr] = acc[v];
        }
    }
    __syncthreads();   // sA/sBv ready; setup overlay region now dead -> z1 bufs

    // ---------------- main loop: 128 M-tiles, double-buffered z1 ------------
    const float bb2 = sb2[wave * 16 + lr];   // b2 fold via WMMA C operand
    float sacc = 0.f;                        // per-lane column sum

    gen_z1(zb0, sA, sBv, sw10, sX, tid, i0, 0);   // prologue: tile 0
    __syncthreads();

    for (int t = 0; t < TI * 8; ++t) {
        _Float16* cur = (t & 1) ? zb1 : zb0;
        _Float16* nxt = (t & 1) ? zb0 : zb1;

        v8f acc;
        #pragma unroll
        for (int v = 0; v < 8; ++v) acc[v] = bb2;
        #pragma unroll
        for (int ks = 0; ks < 4; ++ks) {
            const int ka = ks * 32 + (hi ? 16 : 0);
            v8h alo = *(const v8h*)(cur + lr * HID + ka);
            v8h ahi = *(const v8h*)(cur + lr * HID + ka + 8);
            acc = __builtin_amdgcn_wmma_f32_16x16x32_f16(
                    false, cat8(alo, ahi), false, breg[ks],
                    (short)0, acc, false, false);
        }

        // overlap: produce next tile's z1 while WMMAs drain
        if (t + 1 < TI * 8)
            gen_z1(nxt, sA, sBv, sw10, sX, tid, i0, t + 1);

        #pragma unroll
        for (int v = 0; v < 8; ++v)
            sacc += siluf(acc[v]);

        __syncthreads();   // nxt fully written & cur fully consumed
    }

    // ---------------- deterministic in-block reduction -> workspace ----------
    sRed[wave * 32 + lane] = sacc;
    __syncthreads();
    if (tid < HID) {                      // column n == tid
        const int w = tid >> 4, l = tid & 15;
        partial[blockIdx.x * HID + tid] = sRed[w * 32 + l] + sRed[w * 32 + l + 16];
    }
}

// out[b][o] = (sum_wg s[b])·W3 + N*N*b3   — tiny, 2048 threads
__global__ void sake_finalize_kernel(const float* __restrict__ partial,
                                     const float* __restrict__ W3,
                                     const float* __restrict__ b3,
                                     float* __restrict__ out)
{
    const int t = blockIdx.x * blockDim.x + threadIdx.x;
    if (t >= NB * OUTD) return;
    const int b = t >> 6, o = t & 63;
    float r = (float)(NN * NN) * b3[o];
    for (int k = 0; k < HID; ++k) {
        float s = 0.f;
        #pragma unroll
        for (int q = 0; q < 8; ++q)
            s += partial[(b * 8 + q) * HID + k];
        r += s * W3[k * OUTD + o];
    }
    out[t] = r;
}

extern "C" void kernel_launch(void* const* d_in, const int* in_sizes, int n_in,
                              void* d_out, int out_size, void* d_ws, size_t ws_size,
                              hipStream_t stream) {
    const float* feat  = (const float*)d_in[0];
    const float* coord = (const float*)d_in[1];
    const float* W1    = (const float*)d_in[2];
    const float* b1    = (const float*)d_in[3];
    const float* W2    = (const float*)d_in[4];
    const float* b2    = (const float*)d_in[5];
    const float* W3    = (const float*)d_in[6];
    const float* b3    = (const float*)d_in[7];
    float* out     = (float*)d_out;
    float* partial = (float*)d_ws;       // 256 * 128 f32 = 128 KB scratch

    (void)in_sizes; (void)n_in; (void)out_size; (void)ws_size;

    sake_main_kernel<<<dim3(NB * 8), dim3(256), SMEM_BYTES, stream>>>(
        feat, coord, W1, b1, W2, b2, partial);
    sake_finalize_kernel<<<dim3((NB * OUTD + 255) / 256), dim3(256), 0, stream>>>(
        partial, W3, b3, out);
}
